// Memory_71536975282329
// MI455X (gfx1250) — matrix operand order repaired and checked by reading, not compiled, not run on hardware
//
#include <hip/hip_runtime.h>
#include <hip/hip_bf16.h>
#include <math.h>

// ---------------------------------------------------------------------------
// 2-layer GRU (B=16, T=4096, IN=H=512) + residual, for MI455X (gfx1250).
//
//  * bf16 WMMA (v_wmma_f32_16x16x32_bf16) for all matmuls, f32 accumulation.
//  * Input-side GEMMs: full-grid WMMA GEMM (multi-wave latency hiding).
//  * Recurrence: persistent kernel, 16 WGs; each WG keeps its 96x512 bf16
//    W_hh slice resident in LDS (~97.5KB of 320KB/WGP) for all 4096 steps.
//    Hidden state: f32 slice in owner's LDS (gate math) + bf16 image in
//    global, broadcast each step via GLOBAL_LOAD_ASYNC_TO_LDS_B128
//    (ASYNCcnt) and a grid-wide release/acquire atomic barrier.
// ---------------------------------------------------------------------------

typedef __attribute__((ext_vector_type(16))) __bf16 bf16x16;
typedef __attribute__((ext_vector_type(8)))  float  f32x8;

union FragU { bf16x16 v; uint4 q[2]; };

#define T_STEPS 4096
#define HID     512
#define G3H     1536   // 3*H
#define BATCH   16
#define NWG     16     // workgroups in the persistent recurrence
#define KPAD    520    // 512 + 8 bf16 pad (16B) to spread LDS banks

__device__ __forceinline__ unsigned short f2bf(float f) {
  unsigned int u = __float_as_uint(f);
  unsigned int r = (u + 0x7FFFu + ((u >> 16) & 1u)) >> 16;   // RNE
  return (unsigned short)r;
}
__device__ __forceinline__ float sigm(float x) { return 1.0f / (1.0f + expf(-x)); }

// Async global->LDS copy of 16B per enabled lane (gfx1250, ASYNCcnt-tracked).
__device__ __forceinline__ void async_copy_b128(unsigned lds_byte_off,
                                                const void* gaddr) {
  asm volatile("global_load_async_to_lds_b128 %0, %1, off"
               :: "v"(lds_byte_off), "v"(gaddr) : "memory");
}
__device__ __forceinline__ void wait_async0() {
  asm volatile("s_wait_asynccnt 0x0" ::: "memory");
}

// ---------------------------------------------------------------- conversions
__global__ void to_bf16_kernel(const float4* __restrict__ src,
                               ushort4* __restrict__ dst, long n4) {
  long i = (long)blockIdx.x * blockDim.x + threadIdx.x;
  long stride = (long)gridDim.x * blockDim.x;
  for (; i < n4; i += stride) {
    float4 f = src[i];
    ushort4 p;
    p.x = f2bf(f.x); p.y = f2bf(f.y); p.z = f2bf(f.z); p.w = f2bf(f.w);
    dst[i] = p;
  }
}

__global__ void init_state_kernel(unsigned short* hb1, unsigned short* hb2,
                                  unsigned int* counters) {
  int i = threadIdx.x;
  for (int j = i; j < BATCH * HID; j += blockDim.x) { hb1[j] = 0; hb2[j] = 0; }
  if (i < 2) counters[i] = 0u;
}

// ------------------------------------------------------- input-side big GEMM
// Out[r, n] = sum_k A[r,k] * W[n,k] + bias[n]   A:[rows,512]bf16  W:[1536,512]bf16
// grid = (rows/128, 1536/64), block = 256 (8 waves); wave -> 16x64 strip.
__global__ __launch_bounds__(256)
void gemm_xg_kernel(const unsigned short* __restrict__ A,
                    const unsigned short* __restrict__ W,
                    const float* __restrict__ bias,
                    float* __restrict__ Out) {
  const int wave = threadIdx.x >> 5;
  const int lane = threadIdx.x & 31;
  const int nl   = lane & 15;      // n (and A-row m) within tile
  const int kh   = lane >> 4;      // K half selector
  const long mbase  = ((long)blockIdx.x * 8 + wave) * 16;
  const int  nbase0 = blockIdx.y * 64;

  const unsigned short* arow  = A + (mbase + nl) * 512;
  const unsigned short* wbase = W + (long)(nbase0 + nl) * 512;

  f32x8 acc[4];
  for (int nt = 0; nt < 4; ++nt) {
    float b = bias[nbase0 + nt * 16 + nl];
    for (int v = 0; v < 8; ++v) acc[nt][v] = b;
  }

#pragma unroll
  for (int kc = 0; kc < 16; ++kc) {               // K = 16 * 32
    FragU Af;
    Af.q[0] = *reinterpret_cast<const uint4*>(arow + kc * 32 + kh * 16);
    Af.q[1] = *reinterpret_cast<const uint4*>(arow + kc * 32 + kh * 16 + 8);
#pragma unroll
    for (int nt = 0; nt < 4; ++nt) {
      const unsigned short* wrow = wbase + (long)nt * 16 * 512 + kc * 32 + kh * 16;
      FragU Bf;
      Bf.q[0] = *reinterpret_cast<const uint4*>(wrow);
      Bf.q[1] = *reinterpret_cast<const uint4*>(wrow + 8);
      acc[nt] = __builtin_amdgcn_wmma_f32_16x16x32_bf16(
          false, Af.v, false, Bf.v, (short)0, acc[nt], false, false);
    }
  }

#pragma unroll
  for (int nt = 0; nt < 4; ++nt)
#pragma unroll
    for (int v = 0; v < 8; ++v)
      Out[(mbase + v + 8 * kh) * (long)G3H + nbase0 + nt * 16 + nl] = acc[nt][v];
}

// ------------------------------------------------------ persistent recurrence
// 16 WGs x 192 threads (6 waves). WG wg owns hidden columns [wg*32, wg*32+32).
// Tile ti = jt*3 + g  (jt in {0,1} local 16-col group, g in {r,z,n}) maps to
// W_hh rows  grow = g*512 + wg*32 + jt*16 + nn.  W slice lives in LDS forever.
__global__ __launch_bounds__(192)
void gru_recurrence_kernel(const float* __restrict__ xg,          // [B*T, 1536]
                           const unsigned short* __restrict__ Whh,// [1536,512] bf16
                           const float* __restrict__ bhh,         // [1536]
                           unsigned short* __restrict__ h_bf16g,  // [16,512] bf16
                           unsigned int* __restrict__ counter,
                           float* __restrict__ y_f32,             // nullable
                           unsigned short* __restrict__ y_bf16,   // nullable
                           const float* __restrict__ res,         // nullable
                           float* __restrict__ out) {             // nullable
  extern __shared__ char smem[];
  unsigned short* w_lds = (unsigned short*)smem;                      // 96 x KPAD
  unsigned short* h_lds = (unsigned short*)(smem + 96 * KPAD * 2);    // 16 x KPAD
  float* stage = (float*)(smem + 96 * KPAD * 2 + 16 * KPAD * 2);      // 6 x 16 x 17
  float* h_own = (float*)(smem + 96 * KPAD * 2 + 16 * KPAD * 2 + 6 * 272 * 4); // 16 x 33

  const unsigned w_lds_off = (unsigned)(unsigned long long)(uintptr_t)w_lds;
  const unsigned h_lds_off = (unsigned)(unsigned long long)(uintptr_t)h_lds;

  const int tid  = threadIdx.x;
  const int wave = tid >> 5;
  const int lane = tid & 31;
  const int wg   = blockIdx.x;
  const int hcol_base = wg * 32;

  // ---- one-time: async-DMA this WG's W_hh slice into LDS (16B chunks)
  for (int i = tid; i < 96 * 64; i += 192) {
    int row = i >> 6, c = i & 63;                 // c: 16B chunk column
    int ti = row >> 4, nn = row & 15;
    int grow = (ti % 3) * 512 + hcol_base + (ti / 3) * 16 + nn;
    async_copy_b128(w_lds_off + (unsigned)((row * KPAD + c * 8) * 2),
                    Whh + (long)grow * 512 + c * 8);
  }
  // ---- zero the f32 hidden-state slice (h0 = 0)
  for (int i = tid; i < 16 * 33; i += 192) h_own[i] = 0.0f;
  wait_async0();

  const int nl = lane & 15, kh = lane >> 4;
  const int ti = wave;                            // 6 waves -> 6 tiles
  const int g  = ti % 3, jt = ti / 3;
  const float bias = bhh[g * 512 + hcol_base + jt * 16 + nl];
  const unsigned short* brow = &w_lds[(ti * 16 + nl) * KPAD];
  const unsigned short* harow = &h_lds[nl * KPAD];

  for (int t = 0; t < T_STEPS; ++t) {
    __syncthreads();
    // ---- async broadcast: bf16 h image (global, written last step) -> LDS
    for (int i = tid; i < BATCH * 64; i += 192) {       // 64 x 16B per row
      int m = i >> 6, c = i & 63;
      async_copy_b128(h_lds_off + (unsigned)((m * KPAD + c * 8) * 2),
                      h_bf16g + m * 512 + c * 8);
    }
    wait_async0();
    __syncthreads();

    // ---- hg tile = h(16x512) @ Whh_slice^T(512x16), bf16 WMMA, f32 accum
    f32x8 acc;
#pragma unroll
    for (int v = 0; v < 8; ++v) acc[v] = bias;
#pragma unroll
    for (int kc = 0; kc < 16; ++kc) {
      FragU Af, Bf;
      Af.q[0] = *reinterpret_cast<const uint4*>(harow + kc * 32 + kh * 16);
      Af.q[1] = *reinterpret_cast<const uint4*>(harow + kc * 32 + kh * 16 + 8);
      Bf.q[0] = *reinterpret_cast<const uint4*>(brow + kc * 32 + kh * 16);
      Bf.q[1] = *reinterpret_cast<const uint4*>(brow + kc * 32 + kh * 16 + 8);
      acc = __builtin_amdgcn_wmma_f32_16x16x32_bf16(
          false, Af.v, false, Bf.v, (short)0, acc, false, false);
    }
#pragma unroll
    for (int v = 0; v < 8; ++v)
      stage[ti * 272 + (v + 8 * kh) * 17 + nl] = acc[v];
    __syncthreads();

    // ---- gates + state update for this WG's 32 columns (512 scalars)
    for (int i = tid; i < 512; i += 192) {
      int m = i & 15, c = i >> 4;                 // m = batch, c = local col
      int ljt = c >> 4, n = c & 15;
      int hcol = hcol_base + c;
      float hr = stage[(ljt * 3 + 0) * 272 + m * 17 + n];
      float hz = stage[(ljt * 3 + 1) * 272 + m * 17 + n];
      float hn = stage[(ljt * 3 + 2) * 272 + m * 17 + n];
      long xrow = (long)m * T_STEPS + t;
      const float* xr3 = xg + xrow * G3H;
      float r = sigm(xr3[hcol] + hr);
      float z = sigm(xr3[512 + hcol] + hz);
      float nn = tanhf(xr3[1024 + hcol] + r * hn);
      float hp = h_own[m * 33 + c];
      float hnew = (1.0f - z) * nn + z * hp;
      h_own[m * 33 + c] = hnew;                   // f32 master (owner only)
      h_bf16g[m * 512 + hcol] = f2bf(hnew);       // bf16 broadcast image
      long yi = xrow * HID + hcol;
      if (y_f32)  y_f32[yi]  = hnew;
      if (y_bf16) y_bf16[yi] = f2bf(hnew);
      if (out)    out[yi]    = hnew + res[yi];    // residual skip
    }

    // ---- grid-wide release/acquire barrier (monotonic counter)
    __threadfence();
    __syncthreads();
    if (tid == 0) {
      unsigned int target = (unsigned int)(t + 1) * NWG;
      __hip_atomic_fetch_add(counter, 1u, __ATOMIC_ACQ_REL, __HIP_MEMORY_SCOPE_AGENT);
      while (__hip_atomic_load(counter, __ATOMIC_ACQUIRE,
                               __HIP_MEMORY_SCOPE_AGENT) < target) {
        __builtin_amdgcn_s_sleep(2);
      }
    }
    __syncthreads();
  }
}

// ---------------------------------------------------------------------------
extern "C" void kernel_launch(void* const* d_in, const int* in_sizes, int n_in,
                              void* d_out, int out_size, void* d_ws, size_t ws_size,
                              hipStream_t stream) {
  const float* x     = (const float*)d_in[0];
  const float* W_ih1 = (const float*)d_in[1];
  const float* W_hh1 = (const float*)d_in[2];
  const float* b_ih1 = (const float*)d_in[3];
  const float* b_hh1 = (const float*)d_in[4];
  const float* W_ih2 = (const float*)d_in[5];
  const float* W_hh2 = (const float*)d_in[6];
  const float* b_ih2 = (const float*)d_in[7];
  const float* b_hh2 = (const float*)d_in[8];
  float* out = (float*)d_out;

  const long ROWS = (long)BATCH * T_STEPS;       // 65536
  char* ws = (char*)d_ws;
  size_t off = 0;
  float* xg = (float*)(ws + off);          off += (size_t)ROWS * G3H * 4;   // 402.7MB
  float* y  = (float*)(ws + off);          off += (size_t)ROWS * HID * 4;   // 134.2MB
  unsigned short* actb = (unsigned short*)(ws + off);
                                           off += (size_t)ROWS * HID * 2;   // 67.1MB
  unsigned short* wb[4];
  for (int i = 0; i < 4; ++i) { wb[i] = (unsigned short*)(ws + off); off += (size_t)G3H * HID * 2; }
  unsigned short* hb1 = (unsigned short*)(ws + off); off += BATCH * HID * 2;
  unsigned short* hb2 = (unsigned short*)(ws + off); off += BATCH * HID * 2;
  unsigned int* counters = (unsigned int*)(ws + off); off += 256;

  // 0) zero h0 images and sync counters (every call: deterministic for replay)
  init_state_kernel<<<1, 256, 0, stream>>>(hb1, hb2, counters);

  // 1) bf16 conversions (vectorized float4 -> ushort4)
  to_bf16_kernel<<<4096, 256, 0, stream>>>((const float4*)x, (ushort4*)actb,
                                           ROWS * HID / 4);
  to_bf16_kernel<<<192, 256, 0, stream>>>((const float4*)W_ih1, (ushort4*)wb[0],
                                          (long)G3H * HID / 4);
  to_bf16_kernel<<<192, 256, 0, stream>>>((const float4*)W_hh1, (ushort4*)wb[1],
                                          (long)G3H * HID / 4);
  to_bf16_kernel<<<192, 256, 0, stream>>>((const float4*)W_ih2, (ushort4*)wb[2],
                                          (long)G3H * HID / 4);
  to_bf16_kernel<<<192, 256, 0, stream>>>((const float4*)W_hh2, (ushort4*)wb[3],
                                          (long)G3H * HID / 4);

  dim3 ggrid(ROWS / 128, G3H / 64);
  const size_t smem = (size_t)96 * KPAD * 2 + 16 * KPAD * 2 + 6 * 272 * 4
                    + 16 * 33 * 4;                                   // 125120B

  // 2) layer 1: xg1 = x @ W_ih1^T + b_ih1 ; recurrence -> y (f32 + bf16)
  gemm_xg_kernel<<<ggrid, 256, 0, stream>>>(actb, wb[0], b_ih1, xg);
  gru_recurrence_kernel<<<NWG, 192, smem, stream>>>(
      xg, wb[1], b_hh1, hb1, counters + 0, y, actb, nullptr, nullptr);

  // 3) layer 2: xg2 = y @ W_ih2^T + b_ih2 ; recurrence -> out = h + y
  gemm_xg_kernel<<<ggrid, 256, 0, stream>>>(actb, wb[2], b_ih2, xg);
  gru_recurrence_kernel<<<NWG, 192, smem, stream>>>(
      xg, wb[3], b_hh2, hb2, counters + 1, nullptr, nullptr, y, out);
}
